// ACDEDecoder_30562987278639
// MI455X (gfx1250) — matrix operand order repaired and checked by reading, not compiled, not run on hardware
//
#include <hip/hip_runtime.h>
#include <hip/hip_bf16.h>

// Problem constants (from reference setup_inputs)
#define Bb   4
#define Pp   8
#define Hh   256
#define Wsp  256
#define Cc   200
#define HID  128
#define HW   (Hh * Wsp)          // 65536
#define BN   (Bb * HW)           // 262144
#define CPAD 224                 // Cc padded to multiple of 32
#define KT1  (CPAD / 32)         // 7  K-tiles, layer 1
#define KT2  (HID / 32)          // 4  K-tiles, layer 2
#define NT   (HID / 16)          // 8  N-tiles (MLP)
#define NT2  (CPAD / 16)         // 14 N-tiles (segment GEMM)
#define YHAT_ELEMS (Bb * Cc * HW)

typedef __attribute__((ext_vector_type(16))) __bf16 v16bf;
typedef __attribute__((ext_vector_type(8)))  float  v8f;

union FragB { uint4 u[2]; v16bf bf; };

__device__ inline v8f v8f_zero() {
    v8f z = {0.f, 0.f, 0.f, 0.f, 0.f, 0.f, 0.f, 0.f};
    return z;
}

// fp32 -> bf16 round-to-nearest-even
__device__ inline unsigned short f2bf(float x) {
    unsigned int u = __float_as_uint(x);
    u += 0x7FFFu + ((u >> 16) & 1u);
    return (unsigned short)(u >> 16);
}

__device__ inline float atomicMaxFloat(float* addr, float val) {
    int* ia = (int*)addr;
    int old = __hip_atomic_load(ia, __ATOMIC_RELAXED, __HIP_MEMORY_SCOPE_AGENT);
    while (__int_as_float(old) < val) {
        int assumed = old;
        old = atomicCAS(ia, assumed, __float_as_int(val));
        if (old == assumed) break;
    }
    return __int_as_float(old);
}

// ---------------------------------------------------------------------------
// Kernel: pack W1 (K padded 200->224) and W2 into per-lane WMMA B-fragment
// order: block (kt,nt) = 32 lanes x 16 consecutive-K bf16 (contiguous/lane).
// Lane L of a B-frag holds column n = nt*16 + (L&15), K = kt*32+(L>>4)*16+j.
// ---------------------------------------------------------------------------
__global__ void pack_weights_kernel(const float* __restrict__ W1,
                                    const float* __restrict__ W2,
                                    unsigned short* __restrict__ W1p,
                                    unsigned short* __restrict__ W2p) {
    const int N1 = KT1 * NT * 32 * 16;   // 28672
    const int N2 = KT2 * NT * 32 * 16;   // 16384
    int e = blockIdx.x * 256 + threadIdx.x;
    if (e < N1) {
        int kt   = e / (NT * 512);
        int r    = e % (NT * 512);
        int nt   = r / 512;
        int lane = (r % 512) / 16;
        int j    = e % 16;
        int n    = nt * 16 + (lane & 15);
        int k    = kt * 32 + (lane >> 4) * 16 + j;
        float v  = (k < Cc) ? W1[k * HID + n] : 0.0f;
        W1p[e] = f2bf(v);
    } else if (e < N1 + N2) {
        int e2   = e - N1;
        int kt   = e2 / (NT * 512);
        int r    = e2 % (NT * 512);
        int nt   = r / 512;
        int lane = (r % 512) / 16;
        int j    = e2 % 16;
        int n    = nt * 16 + (lane & 15);
        int k    = kt * 32 + (lane >> 4) * 16 + j;
        W2p[e2] = f2bf(W2[k * HID + n]);
    }
}

// ---------------------------------------------------------------------------
// Kernel: zero the segment accumulators, set segmax = -inf
// ---------------------------------------------------------------------------
__global__ void init_accum_kernel(float* __restrict__ S, float* __restrict__ denom,
                                  float* __restrict__ counts, float* __restrict__ segmax) {
    int i = blockIdx.x * 256 + threadIdx.x;
    if (i < Pp * Cc) S[i] = 0.0f;
    if (i < Pp) {
        denom[i]  = 0.0f;
        counts[i] = 0.0f;
        segmax[i] = -__builtin_inff();
    }
}

// ---------------------------------------------------------------------------
// Kernel: per-pixel class (argmax of abundances) + 3-layer MLP via WMMA bf16.
// One 256-thread block (8 waves) handles 128 pixels. Each wave owns 16 rows.
// B fragments are software-pipelined (load nt+1 while WMMA nt executes).
// ---------------------------------------------------------------------------
__global__ __launch_bounds__(256, 1)
void mlp_logits_kernel(const float* __restrict__ Y,   const float* __restrict__ A,
                       const float* __restrict__ b1,  const float* __restrict__ b2,
                       const float* __restrict__ W3,  const float* __restrict__ b3,
                       const unsigned short* __restrict__ W1p,
                       const unsigned short* __restrict__ W2p,
                       float* __restrict__ logits, int* __restrict__ cls,
                       float* __restrict__ counts, float* __restrict__ segmax) {
    __shared__ unsigned short A1[128][CPAD];   // Y tile, bf16, K padded
    __shared__ unsigned short H1[128][HID];    // hidden tile, bf16
    __shared__ int clss[128];

    const int tid    = threadIdx.x;
    const int bnBase = blockIdx.x * 128;
    const int b      = bnBase / HW;            // blocks never straddle batch
    const int hwBase = bnBase % HW;

    // ---- stage Y -> bf16 LDS (coalesced: tid&127 walks pixels) ----
    {
        const int i  = tid & 127;
        const int hw = hwBase + i;
        for (int c = tid >> 7; c < CPAD; c += 2) {
            float v = 0.0f;
            if (c < Cc) v = Y[(long)(b * Cc + c) * HW + hw];
            A1[i][c] = f2bf(v);
        }
    }
    // ---- per-pixel argmax class + counts ----
    if (tid < 128) {
        const int hw = hwBase + tid;
        float best = A[(long)(b * Pp) * HW + hw];
        int   bi   = 0;
        for (int p = 1; p < Pp; ++p) {
            float v = A[(long)(b * Pp + p) * HW + hw];
            if (v > best) { best = v; bi = p; }
        }
        clss[tid] = bi;
        cls[bnBase + tid] = bi;
        atomicAdd(&counts[bi], 1.0f);
    }
    __syncthreads();

    const int wave    = tid >> 5;
    const int lane    = tid & 31;
    const int rowBase = wave * 16;
    const int halr    = lane >> 4;    // which K-half / row-half
    const int lcol    = lane & 15;

    // ---- layer 1: [16 x 224] @ [224 x 128] per wave ----
    v8f acc[NT];
    #pragma unroll
    for (int i = 0; i < NT; ++i) acc[i] = v8f_zero();

    for (int kt = 0; kt < KT1; ++kt) {
        FragB fa;
        const unsigned short* ap = &A1[rowBase + lcol][kt * 32 + halr * 16];
        fa.u[0] = *(const uint4*)(ap);
        fa.u[1] = *(const uint4*)(ap + 8);
        const uint4* bp = (const uint4*)(W1p) + ((kt * NT) * 32 + lane) * 2;
        FragB fb[2];
        fb[0].u[0] = bp[0];
        fb[0].u[1] = bp[1];
        #pragma unroll
        for (int nt = 0; nt < NT; ++nt) {
            const int cur = nt & 1;
            if (nt + 1 < NT) {
                const uint4* np = bp + (nt + 1) * 64;   // next N-tile frag
                fb[cur ^ 1].u[0] = np[0];
                fb[cur ^ 1].u[1] = np[1];
            }
            acc[nt] = __builtin_amdgcn_wmma_f32_16x16x32_bf16(
                false, fa.bf, false, fb[cur].bf, (short)0, acc[nt], false, false);
        }
    }
    // bias + relu -> bf16 hidden tile (C/D layout: m = 8*halr + r, n = lcol)
    #pragma unroll
    for (int nt = 0; nt < NT; ++nt) {
        const int col = nt * 16 + lcol;
        const float bb = b1[col];
        #pragma unroll
        for (int r = 0; r < 8; ++r) {
            float v = acc[nt][r] + bb;
            v = v > 0.0f ? v : 0.0f;
            H1[rowBase + halr * 8 + r][col] = f2bf(v);
        }
    }
    __syncthreads();

    // ---- layer 2: [16 x 128] @ [128 x 128] per wave ----
    v8f acc2[NT];
    #pragma unroll
    for (int i = 0; i < NT; ++i) acc2[i] = v8f_zero();

    for (int kt = 0; kt < KT2; ++kt) {
        FragB fa;
        const unsigned short* ap = &H1[rowBase + lcol][kt * 32 + halr * 16];
        fa.u[0] = *(const uint4*)(ap);
        fa.u[1] = *(const uint4*)(ap + 8);
        const uint4* bp = (const uint4*)(W2p) + ((kt * NT) * 32 + lane) * 2;
        FragB fb[2];
        fb[0].u[0] = bp[0];
        fb[0].u[1] = bp[1];
        #pragma unroll
        for (int nt = 0; nt < NT; ++nt) {
            const int cur = nt & 1;
            if (nt + 1 < NT) {
                const uint4* np = bp + (nt + 1) * 64;
                fb[cur ^ 1].u[0] = np[0];
                fb[cur ^ 1].u[1] = np[1];
            }
            acc2[nt] = __builtin_amdgcn_wmma_f32_16x16x32_bf16(
                false, fa.bf, false, fb[cur].bf, (short)0, acc2[nt], false, false);
        }
    }

    // ---- layer 3: relu(h2) @ W3 folded out of C/D registers ----
    float partial[8];
    #pragma unroll
    for (int r = 0; r < 8; ++r) partial[r] = 0.0f;
    #pragma unroll
    for (int nt = 0; nt < NT; ++nt) {
        const int col = nt * 16 + lcol;
        const float bb = b2[col];
        const float w3 = W3[col];
        #pragma unroll
        for (int r = 0; r < 8; ++r) {
            float v = acc2[nt][r] + bb;
            v = v > 0.0f ? v : 0.0f;
            partial[r] += v * w3;
        }
    }
    // reduce over the 16 lanes of each half (masks < 16 stay within half)
    #pragma unroll
    for (int m = 1; m < 16; m <<= 1) {
        #pragma unroll
        for (int r = 0; r < 8; ++r) partial[r] += __shfl_xor(partial[r], m, 32);
    }
    if (lcol == 0) {
        const float bias3 = b3[0];
        #pragma unroll
        for (int r = 0; r < 8; ++r) {
            const int row = rowBase + halr * 8 + r;
            const float lg = partial[r] + bias3;
            logits[bnBase + row] = lg;
            atomicMaxFloat(&segmax[clss[row]], lg);
        }
    }
}

// ---------------------------------------------------------------------------
// Kernel: per-class softmax numerators as a masked GEMM:
//   S[p][c] = sum_k (ex_k * [cls_k == p]) * Y[k][c]   ->  S = WmT(16xK) @ Y(KxCPAD)
// One block = 512 pixels processed as 4 sub-tiles of K=128, accumulated in
// WMMA f32 registers; one global atomicAdd per valid (p,c) per block.
// Yc is staged column-major so B-frags are contiguous 32B LDS reads.
// ---------------------------------------------------------------------------
__global__ __launch_bounds__(256, 1)
void accum_wmma_kernel(const float* __restrict__ Y, const float* __restrict__ logits,
                       const int* __restrict__ cls, const float* __restrict__ segmax,
                       float* __restrict__ S, float* __restrict__ denom) {
    __shared__ unsigned short Yc[CPAD][128];   // Y tile, bf16, [c][k]
    __shared__ unsigned short WmT[16][128];    // masked-ex tile, bf16, [p][k]
    __shared__ float dl[Pp];

    const int tid    = threadIdx.x;
    const int wave   = tid >> 5;
    const int lane   = tid & 31;
    const int halr   = lane >> 4;
    const int lcol   = lane & 15;
    const int bnBase = blockIdx.x * 512;       // 512 | HW, so b is constant
    const int b      = bnBase / HW;
    const int hwBase = bnBase % HW;

    if (tid < Pp) dl[tid] = 0.0f;

    // each wave owns up to two N-tiles of the 16 x CPAD partial result
    const int nt0 = wave;           // < 8, always valid
    const int nt1 = wave + 8;       // valid when < NT2(=14)
    v8f acc0 = v8f_zero();
    v8f acc1 = v8f_zero();

    for (int sub = 0; sub < 4; ++sub) {
        // ---- stage Y column-major + masked-ex tile ----
        {
            const int i  = tid & 127;
            const int hw = hwBase + sub * 128 + i;
            for (int c = tid >> 7; c < CPAD; c += 2) {
                float v = 0.0f;
                if (c < Cc) v = Y[(long)(b * Cc + c) * HW + hw];
                Yc[c][i] = f2bf(v);
            }
        }
        if (tid < 128) {
            const int bn = bnBase + sub * 128 + tid;
            const int cl = cls[bn];
            float sm = segmax[cl];
            if (!(sm > -1e37f)) sm = 0.0f;
            const float ex = __expf(logits[bn] - sm);
            atomicAdd(&dl[cl], ex);
            const unsigned short exb = f2bf(ex);
            #pragma unroll
            for (int p = 0; p < 16; ++p)
                WmT[p][tid] = (p == cl) ? exb : (unsigned short)0;
        }
        __syncthreads();

        // ---- GEMM: 4 K-tiles of 32 pixels each ----
        #pragma unroll
        for (int kt = 0; kt < KT2; ++kt) {
            FragB fa;   // A[m=p][k] = WmT[p][k]
            const unsigned short* ap = &WmT[lcol][kt * 32 + halr * 16];
            fa.u[0] = *(const uint4*)(ap);
            fa.u[1] = *(const uint4*)(ap + 8);
            {   // B frag for nt0: lane holds col n, 16 consecutive k
                FragB fb;
                const unsigned short* bp = &Yc[nt0 * 16 + lcol][kt * 32 + halr * 16];
                fb.u[0] = *(const uint4*)(bp);
                fb.u[1] = *(const uint4*)(bp + 8);
                acc0 = __builtin_amdgcn_wmma_f32_16x16x32_bf16(
                    false, fa.bf, false, fb.bf, (short)0, acc0, false, false);
            }
            if (nt1 < NT2) {
                FragB fb;
                const unsigned short* bp = &Yc[nt1 * 16 + lcol][kt * 32 + halr * 16];
                fb.u[0] = *(const uint4*)(bp);
                fb.u[1] = *(const uint4*)(bp + 8);
                acc1 = __builtin_amdgcn_wmma_f32_16x16x32_bf16(
                    false, fa.bf, false, fb.bf, (short)0, acc1, false, false);
            }
        }
        __syncthreads();   // protect Yc/WmT before restaging
    }

    // ---- flush partials: D element (m = 8*halr + r, n) ; only p = m < 8 valid ----
    if (halr == 0) {
        #pragma unroll
        for (int r = 0; r < 8; ++r) {
            const int p = r;
            const int c0 = nt0 * 16 + lcol;
            if (c0 < Cc) atomicAdd(&S[p * Cc + c0], acc0[r]);
            if (nt1 < NT2) {
                const int c1 = nt1 * 16 + lcol;
                if (c1 < Cc) atomicAdd(&S[p * Cc + c1], acc1[r]);
            }
        }
    }
    if (tid < Pp) atomicAdd(&denom[tid], dl[tid]);
}

// ---------------------------------------------------------------------------
// Kernel: finalize M_constrained [C][P] into the tail of d_out
// ---------------------------------------------------------------------------
__global__ void finalize_M_kernel(const float* __restrict__ M, const float* __restrict__ S,
                                  const float* __restrict__ denom,
                                  const float* __restrict__ counts,
                                  float* __restrict__ Mout) {
    const int i = blockIdx.x * 256 + threadIdx.x;
    if (i >= Cc * Pp) return;
    const int c = i / Pp;
    const int p = i % Pp;
    float mb = M[c * Pp + p];
    mb = mb > 0.0f ? mb : 0.0f;
    float v = mb;
    if (counts[p] > 10.0f) {
        float d = denom[p];
        d = d > 1e-30f ? d : 1e-30f;
        const float E = S[p * Cc + c] / d;
        v = 0.5f * E + 0.5f * mb;
    }
    v = v < 0.0f ? 0.0f : (v > 2.0f ? 2.0f : v);
    Mout[i] = v;
}

// ---------------------------------------------------------------------------
// Kernel: Y_hat = A_flat @ M_constrained^T  (bandwidth bound, coalesced stores)
// ---------------------------------------------------------------------------
__global__ __launch_bounds__(256)
void recon_kernel(const float* __restrict__ A, const float* __restrict__ Mc,
                  float* __restrict__ Yhat) {
    __shared__ float Ms[Cc * Pp];
    const int tid = threadIdx.x;
    for (int i = tid; i < Cc * Pp; i += 256) Ms[i] = Mc[i];
    __syncthreads();

    const int bn = blockIdx.x * 256 + tid;
    const int b  = bn / HW;
    const int hw = bn % HW;
    float a[Pp];
    #pragma unroll
    for (int p = 0; p < Pp; ++p) a[p] = A[(long)(b * Pp + p) * HW + hw];
    for (int c = 0; c < Cc; ++c) {
        float s = 0.0f;
        #pragma unroll
        for (int p = 0; p < Pp; ++p) s += a[p] * Ms[c * Pp + p];
        Yhat[(long)(b * Cc + c) * HW + hw] = s;
    }
}

// ---------------------------------------------------------------------------
extern "C" void kernel_launch(void* const* d_in, const int* in_sizes, int n_in,
                              void* d_out, int out_size, void* d_ws, size_t ws_size,
                              hipStream_t stream) {
    const float* A  = (const float*)d_in[0];   // abundances [4,8,256,256]
    const float* Y  = (const float*)d_in[1];   // Y          [4,200,256,256]
    const float* M  = (const float*)d_in[2];   // M          [200,8]
    const float* W1 = (const float*)d_in[3];   // [200,128]
    const float* b1 = (const float*)d_in[4];   // [128]
    const float* W2 = (const float*)d_in[5];   // [128,128]
    const float* b2 = (const float*)d_in[6];   // [128]
    const float* W3 = (const float*)d_in[7];   // [128,1]
    const float* b3 = (const float*)d_in[8];   // [1]

    float* Yhat = (float*)d_out;
    float* Mout = (float*)d_out + YHAT_ELEMS;

    // workspace carve-up
    char* ws = (char*)d_ws;
    unsigned short* W1p = (unsigned short*)ws;               ws += (size_t)KT1 * NT * 512 * 2; // 57344 B
    unsigned short* W2p = (unsigned short*)ws;               ws += (size_t)KT2 * NT * 512 * 2; // 32768 B
    float* logits = (float*)ws;                              ws += (size_t)BN * 4;
    int*   cls    = (int*)ws;                                ws += (size_t)BN * 4;
    float* S      = (float*)ws;                              ws += (size_t)Pp * Cc * 4;
    float* denom  = (float*)ws;                              ws += (size_t)Pp * 4;
    float* counts = (float*)ws;                              ws += (size_t)Pp * 4;
    float* segmax = (float*)ws;                              ws += (size_t)Pp * 4;

    const int NPACK = (KT1 + KT2) * NT * 512;                // 45056
    pack_weights_kernel<<<(NPACK + 255) / 256, 256, 0, stream>>>(W1, W2, W1p, W2p);
    init_accum_kernel<<<(Pp * Cc + 255) / 256, 256, 0, stream>>>(S, denom, counts, segmax);
    mlp_logits_kernel<<<BN / 128, 256, 0, stream>>>(Y, A, b1, b2, W3, b3,
                                                    W1p, W2p, logits, cls, counts, segmax);
    accum_wmma_kernel<<<BN / 512, 256, 0, stream>>>(Y, logits, cls, segmax, S, denom);
    finalize_M_kernel<<<(Cc * Pp + 255) / 256, 256, 0, stream>>>(M, S, denom, counts, Mout);
    recon_kernel<<<BN / 256, 256, 0, stream>>>(A, Mout, Yhat);
}
